// RelativeAttention_9766755631106
// MI455X (gfx1250) — matrix operand order repaired and checked by reading, not compiled, not run on hardware
//
#include <hip/hip_runtime.h>
#include <hip/hip_bf16.h>

// ---------------------------------------------------------------------------
// Relative-position causal attention for MI455X (gfx1250, wave32, WMMA).
// All GEMM-shaped work runs on v_wmma_f32_16x16x32_bf16 (bf16 in, f32 acc).
// Problem: B=4, L=1024, H=1024, NH=16, HD=64, P=2048.
//
// Roofline: ~60 GFLOP total vs ~90 MB unique data -> everything is L2-resident
// (192 MB), HBM time ~4 us @ 23.3 TB/s. Compute-bound, so bf16 WMMA with f32
// accumulation is the right precision/throughput point.
// ---------------------------------------------------------------------------

typedef __bf16 bf16;
typedef __attribute__((ext_vector_type(16))) __bf16 v16bf;
typedef __attribute__((ext_vector_type(8)))  float  v8f;

struct U4x2 { uint4 lo, hi; };
union FragU { v16bf v; U4x2 q; };

static constexpr int kB  = 4;
static constexpr int kL  = 1024;
static constexpr int kH  = 1024;
static constexpr int kNH = 16;
static constexpr int kP  = 2048;
static constexpr int kHD = 64;
static constexpr int kH3 = 3 * kH;
static constexpr int kTD = 1024;           // T2 LDS row length (max rel distance)

__device__ __forceinline__ v8f wmma_bf16(v16bf a, v16bf b, v8f c) {
  // 8 args: (neg_a, A, neg_b, B, c_mod, C, reuse_a, reuse_b)
  return __builtin_amdgcn_wmma_f32_16x16x32_bf16(false, a, false, b, (short)0, c,
                                                 false, false);
}

// A-matrix fragment (16x32 bf16, row-major source with leading dim lda):
// lane<16 holds K=0..7,16..23 of row M=lane; lane>=16 holds K=8..15,24..31.
__device__ __forceinline__ v16bf load_fragA(const bf16* base, int row, int lda, int k0) {
  const int lane = threadIdx.x & 31;
  const bf16* p = base + (size_t)row * lda + k0 + ((lane >> 4) & 1) * 8;
  FragU u;
  u.q.lo = *reinterpret_cast<const uint4*>(p);
  u.q.hi = *reinterpret_cast<const uint4*>(p + 16);
  return u.v;
}

// B-matrix fragment (32x16 bf16). Source is B^T row-major (i.e. N x K rows),
// row selects the N index; lane<16 holds K=0..7,16..23; lane>=16 holds K=8..15,24..31.
__device__ __forceinline__ v16bf load_fragB(const bf16* base, int row, int lda, int k0) {
  const int lane = threadIdx.x & 31;
  const bf16* p = base + (size_t)row * lda + k0 + ((lane >> 4) & 1) * 8;
  FragU u;
  u.q.lo = *reinterpret_cast<const uint4*>(p);
  u.q.hi = *reinterpret_cast<const uint4*>(p + 16);
  return u.v;
}

__device__ __forceinline__ float red_max16(float x) {
#pragma unroll
  for (int m = 1; m < 16; m <<= 1) x = fmaxf(x, __shfl_xor(x, m, 32));
  return x;
}
__device__ __forceinline__ float red_sum16(float x) {
#pragma unroll
  for (int m = 1; m < 16; m <<= 1) x += __shfl_xor(x, m, 32);
  return x;
}

// ---------------------------------------------------------------------------
// fp32 -> bf16 conversion
// ---------------------------------------------------------------------------
__global__ __launch_bounds__(256) void cvt_f32_bf16(const float* __restrict__ src,
                                                    bf16* __restrict__ dst, int n) {
  int i = blockIdx.x * 256 + threadIdx.x;
  if (i < n) dst[i] = (bf16)src[i];
}

// Ersub[n][d][h] = Er[n][(P - d) & (P-1)][h]   (only d in [0,L) is reachable)
__global__ __launch_bounds__(256) void build_ersub(const float* __restrict__ Er,
                                                   bf16* __restrict__ ersub) {
  int i = blockIdx.x * 256 + threadIdx.x;            // total kNH*kL*kHD
  int h = i & (kHD - 1);
  int d = (i >> 6) & (kL - 1);
  int n = i >> 16;
  int src_row = (kP - d) & (kP - 1);
  ersub[i] = (bf16)Er[((size_t)n * kP + src_row) * kHD + h];
}

// Vt[((b*NH+n)*HD+h)*L + l] = qkv[(b*L+l)*3H + 2H + n*HD + h]
__global__ __launch_bounds__(256) void transpose_v(const bf16* __restrict__ qkvb,
                                                   bf16* __restrict__ vtb) {
  int i = blockIdx.x * 256 + threadIdx.x;            // total kB*kNH*kHD*kL
  int l = i & (kL - 1);
  int h = (i >> 10) & (kHD - 1);
  int n = (i >> 16) & (kNH - 1);
  int b = i >> 20;
  vtb[i] = qkvb[((size_t)(b * kL + l)) * kH3 + 2 * kH + n * kHD + h];
}

// ---------------------------------------------------------------------------
// C[M x N] = A[M x K] @ B[N x K]^T + bias ;  one wave per 32x64 tile.
// All 6 fragments of a K=32 slab are preloaded into distinct registers so the
// loads issue as one clause and the 8 WMMAs run without per-op loadcnt waits;
// K-loop unrolled x2 so the next slab's loads overlap this slab's WMMAs.
// ---------------------------------------------------------------------------
__global__ __launch_bounds__(32) void gemm_bf16_nt(const bf16* __restrict__ A,
                                                   const bf16* __restrict__ B,
                                                   const float* __restrict__ bias,
                                                   void* __restrict__ Cptr,
                                                   int N, int K, int bf16out) {
  const int lane = threadIdx.x & 31;
  const int col  = lane & 15;
  const int n0   = blockIdx.x * 64;
  const int m0   = blockIdx.y * 32;

  const bf16* arow0 = A + (size_t)(m0 + col) * K;
  const bf16* arow1 = A + (size_t)(m0 + 16 + col) * K;
  const bf16* brow0 = B + (size_t)(n0 + 0 * 16 + col) * K;
  const bf16* brow1 = B + (size_t)(n0 + 1 * 16 + col) * K;
  const bf16* brow2 = B + (size_t)(n0 + 2 * 16 + col) * K;
  const bf16* brow3 = B + (size_t)(n0 + 3 * 16 + col) * K;

  v8f acc[2][4] = {};
#pragma unroll 2
  for (int k0 = 0; k0 < K; k0 += 32) {
    // one clause of 12 global_load_b128, single wait before the WMMA burst
    v16bf a0 = load_fragA(arow0, 0, 0, k0);
    v16bf a1 = load_fragA(arow1, 0, 0, k0);
    v16bf b0 = load_fragB(brow0, 0, 0, k0);
    v16bf b1 = load_fragB(brow1, 0, 0, k0);
    v16bf b2 = load_fragB(brow2, 0, 0, k0);
    v16bf b3 = load_fragB(brow3, 0, 0, k0);
    if (k0 + 64 < K) {  // pull the slab-after-next toward L2/L0
      __builtin_prefetch(arow0 + k0 + 64, 0, 1);
      __builtin_prefetch(brow0 + k0 + 64, 0, 1);
    }
    acc[0][0] = wmma_bf16(a0, b0, acc[0][0]);
    acc[1][0] = wmma_bf16(a1, b0, acc[1][0]);
    acc[0][1] = wmma_bf16(a0, b1, acc[0][1]);
    acc[1][1] = wmma_bf16(a1, b1, acc[1][1]);
    acc[0][2] = wmma_bf16(a0, b2, acc[0][2]);
    acc[1][2] = wmma_bf16(a1, b2, acc[1][2]);
    acc[0][3] = wmma_bf16(a0, b3, acc[0][3]);
    acc[1][3] = wmma_bf16(a1, b3, acc[1][3]);
  }

  const int half = (lane >> 4) & 1;
#pragma unroll
  for (int mi = 0; mi < 2; ++mi)
#pragma unroll
    for (int t = 0; t < 4; ++t)
#pragma unroll
      for (int r = 0; r < 8; ++r) {
        int row = m0 + mi * 16 + r + half * 8;
        int c   = n0 + t * 16 + col;
        float v = acc[mi][t][r] + bias[c];
        if (bf16out) ((bf16*)Cptr)[(size_t)row * N + c] = (bf16)v;
        else         ((float*)Cptr)[(size_t)row * N + c] = v;
      }
}

// ---------------------------------------------------------------------------
// Fused flash-style causal attention with relative positions.
// Grid: (L/16, NH, B); one wave32 per 16 query rows of one head.
//   Phase A: T2[m][d] = q . Ersub[d] via WMMA into LDS (skewed rel scores).
//   Phase B: stream 32-key chunks; Q@K^T WMMA, per-lane ds_load skew gather,
//            causal mask by select (EXEC stays all-ones), online softmax with
//            __shfl_xor row reductions, P re-layout through LDS, P@V WMMA.
// ---------------------------------------------------------------------------
__global__ __launch_bounds__(32) void attn_rel_kernel(const bf16* __restrict__ qkvb,
                                                      const bf16* __restrict__ vtb,
                                                      const bf16* __restrict__ ersub,
                                                      bf16* __restrict__ attnb) {
  __shared__ bf16 T2s[16 * kTD];   // 32 KB: rel scores T2[m][d] for this row block
  __shared__ bf16 Pbuf[16 * 32];   //  1 KB: probability tile re-layout buffer

  const int lane = threadIdx.x & 31;
  const int col  = lane & 15;
  const int half = (lane >> 4) & 1;
  const int i0   = blockIdx.x * 16;
  const int n    = blockIdx.y;
  const int b    = blockIdx.z;

  const bf16* qbase = qkvb + (size_t)(b * kL) * kH3 + n * kHD;
  const bf16* kbase = qkvb + (size_t)(b * kL) * kH3 + kH + n * kHD;
  const bf16* vbase = vtb  + (size_t)((b * kNH + n) * kHD) * kL;
  const bf16* ebase = ersub + (size_t)n * kTD * kHD;

  // Q fragments (16 rows x 64, two K=32 slabs) stay resident all kernel.
  v16bf qf0 = load_fragA(qbase, i0 + col, kH3, 0);
  v16bf qf1 = load_fragA(qbase, i0 + col, kH3, 32);

  // ---- Phase A: T2[m][d] = q_(i0+m) . Ersub[d] for d in [0, i0+16) ----------
  const int dmax = i0 + 16;
  for (int d0 = 0; d0 < dmax; d0 += 16) {
    v16bf e0 = load_fragB(ebase, d0 + col, kHD, 0);
    v16bf e1 = load_fragB(ebase, d0 + col, kHD, 32);
    v8f t = {};
    t = wmma_bf16(qf0, e0, t);
    t = wmma_bf16(qf1, e1, t);
#pragma unroll
    for (int r = 0; r < 8; ++r) {
      int m = r + half * 8;
      T2s[m * kTD + d0 + col] = (bf16)t[r];
    }
  }
  asm volatile("s_wait_dscnt 0" ::: "memory");   // same-wave LDS RAW

  // ---- Phase B: stream key chunks of 32, online softmax, P@V ---------------
  float m_run[8], l_run[8];
#pragma unroll
  for (int r = 0; r < 8; ++r) { m_run[r] = -3.0e38f; l_run[r] = 0.0f; }
  v8f o0 = {}, o1 = {}, o2 = {}, o3 = {};
  const float scale = 0.125f;                     // 1/sqrt(64)

  const int jend = i0 + 16;
  for (int j0 = 0; j0 < jend; j0 += 32) {
    // S = Q @ K^T for 32 keys (two 16x16 tiles); V frags prefetched alongside
    v16bf kf00 = load_fragB(kbase + (size_t)j0 * kH3,        col, kH3, 0);
    v16bf kf01 = load_fragB(kbase + (size_t)j0 * kH3,        col, kH3, 32);
    v16bf kf10 = load_fragB(kbase + (size_t)(j0 + 16) * kH3, col, kH3, 0);
    v16bf kf11 = load_fragB(kbase + (size_t)(j0 + 16) * kH3, col, kH3, 32);
    v16bf vf0  = load_fragB(vbase, 0 * 16 + col, kL, j0);
    v16bf vf1  = load_fragB(vbase, 1 * 16 + col, kL, j0);
    v16bf vf2  = load_fragB(vbase, 2 * 16 + col, kL, j0);
    v16bf vf3  = load_fragB(vbase, 3 * 16 + col, kL, j0);
    v8f s0 = {}, s1 = {};
    s0 = wmma_bf16(qf0, kf00, s0); s0 = wmma_bf16(qf1, kf01, s0);
    s1 = wmma_bf16(qf0, kf10, s1); s1 = wmma_bf16(qf1, kf11, s1);

#pragma unroll
    for (int r = 0; r < 8; ++r) {
      const int mloc = r + half * 8;
      const int i    = i0 + mloc;
      const int ja   = j0 + col;
      const int jb   = j0 + 16 + col;
      const bool va  = (ja <= i);
      const bool vb  = (jb <= i);
      const int da   = va ? (i - ja) : 0;         // clamp: keep loads in-range,
      const int db   = vb ? (i - jb) : 0;         // select result afterwards
      float ta = (float)T2s[mloc * kTD + da];
      float tb = (float)T2s[mloc * kTD + db];
      float sa = va ? (s0[r] * scale + ta) : -3.0e38f;
      float sb = vb ? (s1[r] * scale + tb) : -3.0e38f;

      float mx   = red_max16(fmaxf(sa, sb));
      float mnew = fmaxf(m_run[r], mx);
      float alpha = __expf(m_run[r] - mnew);
      float pa = __expf(sa - mnew);
      float pb = __expf(sb - mnew);
      float rs = red_sum16(pa + pb);
      l_run[r] = l_run[r] * alpha + rs;
      m_run[r] = mnew;
      o0[r] *= alpha; o1[r] *= alpha; o2[r] *= alpha; o3[r] *= alpha;

      Pbuf[mloc * 32 + col]      = (bf16)pa;
      Pbuf[mloc * 32 + 16 + col] = (bf16)pb;
    }
    asm volatile("s_wait_dscnt 0" ::: "memory");  // P stores -> P frag loads

    // O += P(16x32) @ V(32x64); V^T rows are contiguous -> clean B frags.
    v16bf pf = load_fragA((const bf16*)Pbuf, col, 32, 0);
    o0 = wmma_bf16(pf, vf0, o0);
    o1 = wmma_bf16(pf, vf1, o1);
    o2 = wmma_bf16(pf, vf2, o2);
    o3 = wmma_bf16(pf, vf3, o3);
  }

  // ---- Epilogue: normalize, write bf16 [b][l][H] for the output GEMM -------
  bf16* orow = attnb + (size_t)(b * kL + i0) * kH + n * kHD;
#pragma unroll
  for (int r = 0; r < 8; ++r) {
    const int m = r + half * 8;
    const float inv = 1.0f / l_run[r];
    size_t off = (size_t)m * kH;
    orow[off + 0 * 16 + col] = (bf16)(o0[r] * inv);
    orow[off + 1 * 16 + col] = (bf16)(o1[r] * inv);
    orow[off + 2 * 16 + col] = (bf16)(o2[r] * inv);
    orow[off + 3 * 16 + col] = (bf16)(o3[r] * inv);
  }
}

// ---------------------------------------------------------------------------
extern "C" void kernel_launch(void* const* d_in, const int* in_sizes, int n_in,
                              void* d_out, int out_size, void* d_ws, size_t ws_size,
                              hipStream_t stream) {
  const float* x    = (const float*)d_in[0];
  // d_in[1]: causal mask — structure is known statically, unused.
  const float* Wqkv = (const float*)d_in[2];
  const float* bqkv = (const float*)d_in[3];
  const float* Wo   = (const float*)d_in[4];
  const float* bo   = (const float*)d_in[5];
  const float* Er   = (const float*)d_in[6];
  float* out = (float*)d_out;

  char* ws = (char*)d_ws;
  size_t off = 0;
  auto carve = [&](size_t bytes) -> void* {
    void* p = ws + off;
    off += (bytes + 255) & ~(size_t)255;
    return p;
  };
  const size_t nX = (size_t)kB * kL * kH;        // 4M
  bf16* xb    = (bf16*)carve(nX * 2);                       // x bf16
  bf16* wqkvb = (bf16*)carve((size_t)kH3 * kH * 2);         // Wqkv bf16
  bf16* wob   = (bf16*)carve((size_t)kH * kH * 2);          // Wo bf16
  bf16* ersub = (bf16*)carve((size_t)kNH * kTD * kHD * 2);  // gathered Er bf16
  bf16* qkvb  = (bf16*)carve((size_t)kB * kL * kH3 * 2);    // qkv projection
  bf16* vtb   = (bf16*)carve((size_t)kB * kNH * kHD * kL * 2); // V transposed
  bf16* attnb = (bf16*)carve(nX * 2);                       // attention output

  // 1) precision conversion + Er gather
  cvt_f32_bf16<<<(int)((nX + 255) / 256), 256, 0, stream>>>(x, xb, (int)nX);
  cvt_f32_bf16<<<(kH3 * kH + 255) / 256, 256, 0, stream>>>(Wqkv, wqkvb, kH3 * kH);
  cvt_f32_bf16<<<(kH * kH + 255) / 256, 256, 0, stream>>>(Wo, wob, kH * kH);
  build_ersub<<<(kNH * kL * kHD + 255) / 256, 256, 0, stream>>>(Er, ersub);

  // 2) QKV projection: [4096 x 1024] @ [3072 x 1024]^T -> bf16
  gemm_bf16_nt<<<dim3(kH3 / 64, (kB * kL) / 32), 32, 0, stream>>>(
      xb, wqkvb, bqkv, qkvb, kH3, kH, /*bf16out=*/1);

  // 3) V -> [b][h][hd][L] so P@V B-fragments are contiguous loads
  transpose_v<<<(int)(nX / 256), 256, 0, stream>>>(qkvb, vtb);

  // 4) fused relative-position causal attention
  attn_rel_kernel<<<dim3(kL / 16, kNH, kB), 32, 0, stream>>>(qkvb, vtb, ersub, attnb);

  // 5) output projection: [4096 x 1024] @ [1024 x 1024]^T -> fp32 d_out
  gemm_bf16_nt<<<dim3(kH / 64, (kB * kL) / 32), 32, 0, stream>>>(
      attnb, wob, bo, out, kH, kH, /*bf16out=*/0);
}
